// ConditionalWaveNet_34102040330374
// MI455X (gfx1250) — matrix operand order repaired and checked by reading, not compiled, not run on hardware
//
#include <hip/hip_runtime.h>
#include <hip/hip_bf16.h>
#include <math.h>

#define BATCH   8
#define TLEN    8192
#define NLAYERS 40

typedef __attribute__((ext_vector_type(16))) __bf16 v16bf;
typedef __attribute__((ext_vector_type(8)))  __bf16 v8bf;
typedef __attribute__((ext_vector_type(8)))  float  v8f;

__device__ __forceinline__ int lane_id() { return threadIdx.x & 31; }

__device__ __forceinline__ v8f zero8() {
  v8f z = {0.f,0.f,0.f,0.f,0.f,0.f,0.f,0.f};
  return z;
}

__device__ __forceinline__ v8f wmma_bf16(v16bf a, v16bf b, v8f c) {
  return __builtin_amdgcn_wmma_f32_16x16x32_bf16(false, a, false, b, (short)0, c,
                                                 false, false);
}

// ---- fragment builders -----------------------------------------------------
// A fragment (16x32 bf16), M = 16 time rows of a [T][C] f32 activation matrix.
__device__ __forceinline__ v16bf a_frag_f32(const float* __restrict__ p, int row0,
                                            int C, int Tmax) {
  int lane = lane_id();
  int row  = row0 + (lane & 15);
  int kb   = (lane & 16) ? 8 : 0;
  v16bf a;
  if (row >= 0 && row < Tmax) {
    const float* r = p + (size_t)row * C + kb;
#pragma unroll
    for (int j = 0; j < 8; ++j) { a[j] = (__bf16)r[j]; a[8 + j] = (__bf16)r[16 + j]; }
  } else {
#pragma unroll
    for (int j = 0; j < 16; ++j) a[j] = (__bf16)0.f;
  }
  return a;
}

__device__ __forceinline__ v16bf a_frag_bf16(const __bf16* __restrict__ p, int row0,
                                             int C, int cb, int Tmax) {
  int lane = lane_id();
  int row  = row0 + (lane & 15);
  int kb   = cb + ((lane & 16) ? 8 : 0);
  v16bf a;
  if (row >= 0 && row < Tmax) {
    const __bf16* r = p + (size_t)row * C + kb;
#pragma unroll
    for (int j = 0; j < 8; ++j) { a[j] = r[j]; a[8 + j] = r[16 + j]; }
  } else {
#pragma unroll
    for (int j = 0; j < 16; ++j) a[j] = (__bf16)0.f;
  }
  return a;
}

// A fragment from a 16-row f32 LDS tile (rows 0..15).
__device__ __forceinline__ v16bf a_frag_lds(const float* p, int C, int cb) {
  int lane = lane_id();
  int row  = lane & 15;
  int kb   = cb + ((lane & 16) ? 8 : 0);
  const float* r = p + row * C + kb;
  v16bf a;
#pragma unroll
  for (int j = 0; j < 8; ++j) { a[j] = (__bf16)r[j]; a[8 + j] = (__bf16)r[16 + j]; }
  return a;
}

// Pre-packed B fragment: 32 lanes x 16 bf16, lane-contiguous (32B per lane).
__device__ __forceinline__ v16bf b_frag_packed(const __bf16* __restrict__ pk, int f) {
  return *reinterpret_cast<const v16bf*>(pk + ((size_t)f * 32 + lane_id()) * 16);
}

// ---- D-tile helpers (C/D layout: VGPR r -> M=r(+8), N=lane&15) -------------
__device__ __forceinline__ void store_tile_bf16(__bf16* __restrict__ dst, v8f acc,
                                                int n0, int C, int t0) {
  int lane = lane_id();
  int col  = n0 + (lane & 15);
  int rb   = (lane & 16) ? 8 : 0;
#pragma unroll
  for (int r = 0; r < 8; ++r) dst[(size_t)(t0 + rb + r) * C + col] = (__bf16)acc[r];
}

__device__ __forceinline__ void stage_tile(float* lds, v8f acc, int n0, int C) {
  int lane = lane_id();
  int col  = n0 + (lane & 15);
  int rb   = (lane & 16) ? 8 : 0;
#pragma unroll
  for (int r = 0; r < 8; ++r) lds[(rb + r) * C + col] = acc[r];
}

__device__ __forceinline__ void lds_fence() {
  asm volatile("s_wait_dscnt 0" ::: "memory");
}

// ---- weight pre-pack: f32 [L][oc][ic(,tap)] -> bf16 B fragments ------------
// fragment order per layer: ((tap*ktiles + kt)*ntiles + nt)
__global__ void k_packB(const float* __restrict__ src, __bf16* __restrict__ dst,
                        int total_frags, int layer_stride, int frags_per_layer,
                        int ktiles, int ntiles, int oc_stride, int k_stride) {
  int tid = blockIdx.x * blockDim.x + threadIdx.x;
  if (tid >= total_frags * 32) return;
  int lane = tid & 31;
  int f    = tid >> 5;
  int l    = f / frags_per_layer;
  int fl   = f % frags_per_layer;
  int nt   = fl % ntiles;
  int rest = fl / ntiles;
  int kt   = rest % ktiles;
  int tap  = rest / ktiles;
  int oc = nt * 16 + (lane & 15);
  int kb = kt * 32 + ((lane & 16) ? 8 : 0);
  const float* s = src + (size_t)l * layer_stride + tap + (size_t)oc * oc_stride;
  __bf16* d = dst + ((size_t)f * 32 + lane) * 16;
#pragma unroll
  for (int j = 0; j < 16; ++j) d[j] = (__bf16)s[(size_t)(kb + j + (j >= 8 ? 8 : 0)) * k_stride];
}

// ---- conditioning layer 1: 1->32 k7 pad3, BN, ReLU (tiny: VALU) ------------
__global__ void k_cond1(const float* __restrict__ x, const float* __restrict__ w,
                        const float* __restrict__ bias, const float* __restrict__ bnp,
                        __bf16* __restrict__ c1) {
  int idx = blockIdx.x * blockDim.x + threadIdx.x;
  if (idx >= BATCH * TLEN) return;
  int b = idx / TLEN, t = idx % TLEN;
  const float* xb = x + (size_t)b * TLEN;
  float xv[7];
#pragma unroll
  for (int j = 0; j < 7; ++j) {
    int tt = t + j - 3;
    xv[j] = (tt >= 0 && tt < TLEN) ? xb[tt] : 0.f;
  }
  __bf16* out = c1 + (size_t)idx * 32;
#pragma unroll
  for (int oc = 0; oc < 32; ++oc) {
    float s = bias[oc];
#pragma unroll
    for (int j = 0; j < 7; ++j) s += w[oc * 7 + j] * xv[j];
    float sc = bnp[0 * 32 + oc] * rsqrtf(bnp[3 * 32 + oc] + 1e-5f);
    float sh = bnp[1 * 32 + oc] - bnp[2 * 32 + oc] * sc;
    out[oc] = (__bf16)fmaxf(s * sc + sh, 0.f);
  }
}

// ---- start conv: 1->32 k1 (VALU) -------------------------------------------
__global__ void k_start(const float* __restrict__ x, const float* __restrict__ w,
                        const float* __restrict__ bias, float* __restrict__ h) {
  int idx = blockIdx.x * blockDim.x + threadIdx.x;
  if (idx >= BATCH * TLEN) return;
  float xv = x[idx];
  float* out = h + (size_t)idx * 32;
#pragma unroll
  for (int oc = 0; oc < 32; ++oc) out[oc] = w[oc] * xv + bias[oc];
}

// ---- conditioning layer 2: 32->64 k5 pad2 (WMMA) ---------------------------
__global__ void k_cond2(const __bf16* __restrict__ c1, const __bf16* __restrict__ pk,
                        const float* __restrict__ bias, const float* __restrict__ bnp,
                        __bf16* __restrict__ c2) {
  int gw = (blockIdx.x * blockDim.x + threadIdx.x) >> 5;
  int b  = gw / (TLEN / 16);
  int t0 = (gw % (TLEN / 16)) * 16;
  int colN = lane_id() & 15;
  const __bf16* src = c1 + (size_t)b * TLEN * 32;
  __bf16* dst = c2 + (size_t)b * TLEN * 64;
  v16bf A[5];
#pragma unroll
  for (int j = 0; j < 5; ++j) A[j] = a_frag_bf16(src, t0 + j - 2, 32, 0, TLEN);
#pragma unroll
  for (int nt = 0; nt < 4; ++nt) {
    v8f acc = zero8();
#pragma unroll
    for (int j = 0; j < 5; ++j)
      acc = wmma_bf16(A[j], b_frag_packed(pk, j * 4 + nt), acc);
    int col = nt * 16 + colN;
    float sc = bnp[0 * 64 + col] * rsqrtf(bnp[3 * 64 + col] + 1e-5f);
    float sh = bnp[1 * 64 + col] - bnp[2 * 64 + col] * sc;
    float bb = bias[col];
#pragma unroll
    for (int r = 0; r < 8; ++r) acc[r] = fmaxf((acc[r] + bb) * sc + sh, 0.f);
    store_tile_bf16(dst, acc, nt * 16, 64, t0);
  }
}

// ---- conditioning layer 3: 64->128 k3 pad1 (WMMA) --------------------------
__global__ void k_cond3(const __bf16* __restrict__ c2, const __bf16* __restrict__ pk,
                        const float* __restrict__ bias, const float* __restrict__ bnp,
                        __bf16* __restrict__ c3) {
  int gw = (blockIdx.x * blockDim.x + threadIdx.x) >> 5;
  int b  = gw / (TLEN / 16);
  int t0 = (gw % (TLEN / 16)) * 16;
  int colN = lane_id() & 15;
  const __bf16* src = c2 + (size_t)b * TLEN * 64;
  __bf16* dst = c3 + (size_t)b * TLEN * 128;
  v16bf A[3][2];
#pragma unroll
  for (int j = 0; j < 3; ++j)
#pragma unroll
    for (int kc = 0; kc < 2; ++kc)
      A[j][kc] = a_frag_bf16(src, t0 + j - 1, 64, kc * 32, TLEN);
#pragma unroll
  for (int nt = 0; nt < 8; ++nt) {
    v8f acc = zero8();
#pragma unroll
    for (int j = 0; j < 3; ++j)
#pragma unroll
      for (int kc = 0; kc < 2; ++kc)
        acc = wmma_bf16(A[j][kc], b_frag_packed(pk, (j * 2 + kc) * 8 + nt), acc);
    int col = nt * 16 + colN;
    float sc = bnp[0 * 128 + col] * rsqrtf(bnp[3 * 128 + col] + 1e-5f);
    float sh = bnp[1 * 128 + col] - bnp[2 * 128 + col] * sc;
    float bb = bias[col];
#pragma unroll
    for (int r = 0; r < 8; ++r) acc[r] = fmaxf((acc[r] + bb) * sc + sh, 0.f);
    store_tile_bf16(dst, acc, nt * 16, 128, t0);
  }
}

// ---- one WaveNet layer, fully fused ----------------------------------------
__global__ void k_layer(const float* __restrict__ h_in, float* __restrict__ h_out,
                        const __bf16* __restrict__ c, __bf16* __restrict__ g_out,
                        const __bf16* __restrict__ pk_dil, const float* __restrict__ dil_b,
                        const __bf16* __restrict__ pk_cond, const float* __restrict__ condc_b,
                        const __bf16* __restrict__ pk_filt, const float* __restrict__ filt_b,
                        const __bf16* __restrict__ pk_gate, const float* __restrict__ gate_b,
                        const __bf16* __restrict__ pk_res, const float* __restrict__ res_b,
                        const float* __restrict__ blk_bn, int layer, int dil) {
  __shared__ float sbuf[8][16 * 32];          // per-wave private staging tile
  float* lbuf = sbuf[threadIdx.x >> 5];
  int gw = (blockIdx.x * blockDim.x + threadIdx.x) >> 5;
  int b  = gw / (TLEN / 16);
  int t0 = (gw % (TLEN / 16)) * 16;
  int lane = lane_id();
  int colN = lane & 15;
  int rb   = (lane & 16) ? 8 : 0;

  const float*  hb = h_in  + (size_t)b * TLEN * 32;
  float*        ho = h_out + (size_t)b * TLEN * 32;
  const __bf16* cb = c     + (size_t)b * TLEN * 128;

  const float* bnp = blk_bn  + (size_t)layer * 128;
  const float* db  = dil_b   + (size_t)layer * 32;
  const float* ccb = condc_b + (size_t)layer * 32;
  const float* fbp = filt_b  + (size_t)layer * 32;
  const float* gbp = gate_b  + (size_t)layer * 32;
  const float* rbp = res_b   + (size_t)layer * 32;

  v16bf Ah0 = a_frag_f32(hb, t0, 32, TLEN);        // h[t]   (tap 1)
  v16bf Ah1 = a_frag_f32(hb, t0 - dil, 32, TLEN);  // h[t-d] (tap 0), zero-pad
  v16bf Ac[4];
#pragma unroll
  for (int kc = 0; kc < 4; ++kc) Ac[kc] = a_frag_bf16(cb, t0, 128, kc * 32, TLEN);

  // comb = dil_conv(h) + cond_conv(c) + biases  (32 out -> 2 N tiles)
#pragma unroll
  for (int nt = 0; nt < 2; ++nt) {
    v8f acc = zero8();
    acc = wmma_bf16(Ah1, b_frag_packed(pk_dil, layer * 4 + 0 * 2 + nt), acc);
    acc = wmma_bf16(Ah0, b_frag_packed(pk_dil, layer * 4 + 1 * 2 + nt), acc);
#pragma unroll
    for (int kc = 0; kc < 4; ++kc)
      acc = wmma_bf16(Ac[kc], b_frag_packed(pk_cond, layer * 8 + kc * 2 + nt), acc);
    int col = nt * 16 + colN;
    float bb = db[col] + ccb[col];
#pragma unroll
    for (int r = 0; r < 8; ++r) acc[r] += bb;
    stage_tile(lbuf, acc, nt * 16, 32);
  }
  lds_fence();
  v16bf Acomb = a_frag_lds(lbuf, 32, 0);
  lds_fence();

  // gated = BN(tanh(filt(comb)) * sigmoid(gate(comb)))
#pragma unroll
  for (int nt = 0; nt < 2; ++nt) {
    v8f f = wmma_bf16(Acomb, b_frag_packed(pk_filt, layer * 2 + nt), zero8());
    v8f g = wmma_bf16(Acomb, b_frag_packed(pk_gate, layer * 2 + nt), zero8());
    int col = nt * 16 + colN;
    float fbv = fbp[col], gbv = gbp[col];
    float sc = bnp[col] * rsqrtf(bnp[96 + col] + 1e-5f);
    float sh = bnp[32 + col] - bnp[64 + col] * sc;
    v8f gd;
#pragma unroll
    for (int r = 0; r < 8; ++r) {
      float tv = tanhf(f[r] + fbv);
      float sv = 1.f / (1.f + expf(-(g[r] + gbv)));
      gd[r] = (tv * sv) * sc + sh;
    }
    stage_tile(lbuf, gd, nt * 16, 32);
  }
  lds_fence();
  v16bf Ag = a_frag_lds(lbuf, 32, 0);

  // persist gated (bf16, time-major) for the deferred skip GEMM
  {
    int row = t0 + (lane & 15);
    int kb  = (lane & 16) ? 8 : 0;
    __bf16* gdst = g_out + ((size_t)b * TLEN + row) * 32;
    v8bf lo, hi;
#pragma unroll
    for (int j = 0; j < 8; ++j) { lo[j] = Ag[j]; hi[j] = Ag[8 + j]; }
    *reinterpret_cast<v8bf*>(gdst + kb)      = lo;
    *reinterpret_cast<v8bf*>(gdst + kb + 16) = hi;
  }

  // h_out = res_conv(gated) + bias + h_in
#pragma unroll
  for (int nt = 0; nt < 2; ++nt) {
    v8f rs = wmma_bf16(Ag, b_frag_packed(pk_res, layer * 2 + nt), zero8());
    int col = nt * 16 + colN;
    float bb = rbp[col];
#pragma unroll
    for (int r = 0; r < 8; ++r) {
      size_t idx = (size_t)(t0 + rb + r) * 32 + col;
      ho[idx] = rs[r] + bb + hb[idx];
    }
  }
}

// ---- head A: skip GEMM (K=1280) + ReLU + end1 + ReLU -> y1 (bf16) ----------
// 32-time tile per wave-group: both A row-groups share every B fragment.
__global__ void k_headA(const __bf16* __restrict__ gall, const __bf16* __restrict__ pk_skip,
                        const float* __restrict__ skip_b, const __bf16* __restrict__ pk_e1,
                        const float* __restrict__ e1b, __bf16* __restrict__ y1) {
  __shared__ float ybuf[32 * 256];
  int bid = blockIdx.x;
  int b   = bid / (TLEN / 32);
  int t0  = (bid % (TLEN / 32)) * 32;
  int wq  = threadIdx.x >> 5;                // wave -> 64 of 256 out channels
  int colN = lane_id() & 15;

  v8f acc[2][4];
#pragma unroll
  for (int h = 0; h < 2; ++h)
#pragma unroll
    for (int nt = 0; nt < 4; ++nt) acc[h][nt] = zero8();

  for (int i = 0; i < NLAYERS; ++i) {
    const __bf16* g = gall + ((size_t)i * BATCH + b) * TLEN * 32;
    if (i + 1 < NLAYERS)
      __builtin_prefetch(gall + ((size_t)(i + 1) * BATCH + b) * TLEN * 32 +
                         (size_t)t0 * 32, 0, 1);
    v16bf A0 = a_frag_bf16(g, t0, 32, 0, TLEN);
    v16bf A1 = a_frag_bf16(g, t0 + 16, 32, 0, TLEN);
#pragma unroll
    for (int nt = 0; nt < 4; ++nt) {
      v16bf Bf = b_frag_packed(pk_skip, i * 16 + wq * 4 + nt);
      acc[0][nt] = wmma_bf16(A0, Bf, acc[0][nt]);
      acc[1][nt] = wmma_bf16(A1, Bf, acc[1][nt]);
    }
  }
#pragma unroll
  for (int h = 0; h < 2; ++h)
#pragma unroll
    for (int nt = 0; nt < 4; ++nt) {
      int col = wq * 64 + nt * 16 + colN;
      float bb = skip_b[col];
      v8f t = acc[h][nt];
#pragma unroll
      for (int r = 0; r < 8; ++r) t[r] = fmaxf(t[r] + bb, 0.f);
      stage_tile(ybuf + h * 16 * 256, t, wq * 64 + nt * 16, 256);
    }
  __syncthreads();

  v8f acc2[2][4];
#pragma unroll
  for (int h = 0; h < 2; ++h)
#pragma unroll
    for (int nt = 0; nt < 4; ++nt) acc2[h][nt] = zero8();
  for (int kc = 0; kc < 8; ++kc) {
    v16bf Ay0 = a_frag_lds(ybuf, 256, kc * 32);
    v16bf Ay1 = a_frag_lds(ybuf + 16 * 256, 256, kc * 32);
#pragma unroll
    for (int nt = 0; nt < 4; ++nt) {
      v16bf Bf = b_frag_packed(pk_e1, kc * 16 + wq * 4 + nt);
      acc2[0][nt] = wmma_bf16(Ay0, Bf, acc2[0][nt]);
      acc2[1][nt] = wmma_bf16(Ay1, Bf, acc2[1][nt]);
    }
  }
  __bf16* dst = y1 + (size_t)b * TLEN * 256;
#pragma unroll
  for (int h = 0; h < 2; ++h)
#pragma unroll
    for (int nt = 0; nt < 4; ++nt) {
      int col = wq * 64 + nt * 16 + colN;
      float bb = e1b[col];
#pragma unroll
      for (int r = 0; r < 8; ++r) acc2[h][nt][r] = fmaxf(acc2[h][nt][r] + bb, 0.f);
      store_tile_bf16(dst, acc2[h][nt], wq * 64 + nt * 16, 256, t0 + h * 16);
    }
}

// ---- head B: end2 + ReLU, pred, unc1 + ReLU, unc2 + softplus ---------------
__global__ void k_headB(const __bf16* __restrict__ y1, const __bf16* __restrict__ pk_e2,
                        const float* __restrict__ e2b, const float* __restrict__ predw,
                        const float* __restrict__ predb, const __bf16* __restrict__ pk_u1,
                        const float* __restrict__ u1b, const float* __restrict__ u2w,
                        const float* __restrict__ u2b, float* __restrict__ out) {
  __shared__ float y2buf[16 * 256];
  __shared__ float ubuf[16 * 128];
  int bid = blockIdx.x;
  int b   = bid / (TLEN / 16);
  int t0  = (bid % (TLEN / 16)) * 16;
  int wq  = threadIdx.x >> 5;
  int colN = lane_id() & 15;
  const __bf16* src = y1 + (size_t)b * TLEN * 256;

  v8f acc[4];
#pragma unroll
  for (int nt = 0; nt < 4; ++nt) acc[nt] = zero8();
  for (int kc = 0; kc < 8; ++kc) {
    v16bf Ay = a_frag_bf16(src, t0, 256, kc * 32, TLEN);
#pragma unroll
    for (int nt = 0; nt < 4; ++nt)
      acc[nt] = wmma_bf16(Ay, b_frag_packed(pk_e2, kc * 16 + wq * 4 + nt), acc[nt]);
  }
#pragma unroll
  for (int nt = 0; nt < 4; ++nt) {
    int col = wq * 64 + nt * 16 + colN;
    float bb = e2b[col];
    v8f t = acc[nt];
#pragma unroll
    for (int r = 0; r < 8; ++r) t[r] = fmaxf(t[r] + bb, 0.f);
    stage_tile(y2buf, t, wq * 64 + nt * 16, 256);
  }
  __syncthreads();

  // unc1: 256 -> 128, each wave covers 32 out channels (2 N tiles)
  v8f ua[2];
#pragma unroll
  for (int nt = 0; nt < 2; ++nt) ua[nt] = zero8();
  for (int kc = 0; kc < 8; ++kc) {
    v16bf Ay = a_frag_lds(y2buf, 256, kc * 32);
#pragma unroll
    for (int nt = 0; nt < 2; ++nt)
      ua[nt] = wmma_bf16(Ay, b_frag_packed(pk_u1, kc * 8 + wq * 2 + nt), ua[nt]);
  }
#pragma unroll
  for (int nt = 0; nt < 2; ++nt) {
    int col = wq * 32 + nt * 16 + colN;
    float bb = u1b[col];
#pragma unroll
    for (int r = 0; r < 8; ++r) ua[nt][r] = fmaxf(ua[nt][r] + bb, 0.f);
    stage_tile(ubuf, ua[nt], wq * 32 + nt * 16, 128);
  }
  __syncthreads();

  int tid = threadIdx.x;
  if (tid < 16) {                 // predictions: 256 -> 1
    float s = predb[0];
    for (int cc = 0; cc < 256; ++cc) s += predw[cc] * y2buf[tid * 256 + cc];
    out[(size_t)b * TLEN + t0 + tid] = s;
  } else if (tid >= 64 && tid < 80) {  // uncertainty: softplus(128 -> 1)
    int rr = tid - 64;
    float s = u2b[0];
    for (int cc = 0; cc < 128; ++cc) s += u2w[cc] * ubuf[rr * 128 + cc];
    float sp = (s > 0.f) ? (s + log1pf(expf(-s))) : log1pf(expf(s));
    out[(size_t)BATCH * TLEN + (size_t)b * TLEN + t0 + rr] = sp;
  }
}

// ---------------------------------------------------------------------------
extern "C" void kernel_launch(void* const* d_in, const int* in_sizes, int n_in,
                              void* d_out, int out_size, void* d_ws, size_t ws_size,
                              hipStream_t stream) {
  (void)in_sizes; (void)n_in; (void)out_size; (void)ws_size;
  const float* x       = (const float*)d_in[0];
  const float* cond_w1 = (const float*)d_in[1];
  const float* cond_b1 = (const float*)d_in[2];
  const float* cond_bn1= (const float*)d_in[3];
  const float* cond_w2 = (const float*)d_in[4];
  const float* cond_b2 = (const float*)d_in[5];
  const float* cond_bn2= (const float*)d_in[6];
  const float* cond_w3 = (const float*)d_in[7];
  const float* cond_b3 = (const float*)d_in[8];
  const float* cond_bn3= (const float*)d_in[9];
  const float* start_w = (const float*)d_in[10];
  const float* start_b = (const float*)d_in[11];
  const float* dil_w   = (const float*)d_in[12];
  const float* dil_b   = (const float*)d_in[13];
  const float* condc_w = (const float*)d_in[14];
  const float* condc_b = (const float*)d_in[15];
  const float* filt_w  = (const float*)d_in[16];
  const float* filt_b  = (const float*)d_in[17];
  const float* gate_w  = (const float*)d_in[18];
  const float* gate_b  = (const float*)d_in[19];
  const float* skip_w  = (const float*)d_in[20];
  const float* skip_b  = (const float*)d_in[21];
  const float* res_w   = (const float*)d_in[22];
  const float* res_b   = (const float*)d_in[23];
  const float* blk_bn  = (const float*)d_in[24];
  const float* end1_w  = (const float*)d_in[25];
  const float* end1_b  = (const float*)d_in[26];
  const float* end2_w  = (const float*)d_in[27];
  const float* end2_b  = (const float*)d_in[28];
  const float* pred_w  = (const float*)d_in[29];
  const float* pred_b  = (const float*)d_in[30];
  const float* unc_w1  = (const float*)d_in[31];
  const float* unc_b1  = (const float*)d_in[32];
  const float* unc_w2  = (const float*)d_in[33];
  const float* unc_b2  = (const float*)d_in[34];

  const size_t bt = (size_t)BATCH * TLEN;
  char* wp = (char*)d_ws;
  __bf16* c1 = (__bf16*)wp; wp += bt * 32 * 2;                    // 4 MB
  __bf16* c2 = (__bf16*)wp; wp += bt * 64 * 2;                    // 8 MB
  __bf16* c3 = (__bf16*)wp; wp += bt * 128 * 2;                   // 16 MB
  float*  h0 = (float*)wp;  wp += bt * 32 * 4;                    // 8 MB
  float*  h1 = (float*)wp;  wp += bt * 32 * 4;                    // 8 MB
  __bf16* gall = (__bf16*)wp; wp += (size_t)NLAYERS * bt * 32 * 2;// 160 MB
  __bf16* y1 = (__bf16*)wp; wp += bt * 256 * 2;                   // 32 MB
  // packed weight arenas (each fragment = 512 bf16 = 1 KB)
  __bf16* pk_dil  = (__bf16*)wp; wp += (size_t)160 * 512 * 2;
  __bf16* pk_cond = (__bf16*)wp; wp += (size_t)320 * 512 * 2;
  __bf16* pk_filt = (__bf16*)wp; wp += (size_t)80  * 512 * 2;
  __bf16* pk_gate = (__bf16*)wp; wp += (size_t)80  * 512 * 2;
  __bf16* pk_res  = (__bf16*)wp; wp += (size_t)80  * 512 * 2;
  __bf16* pk_skip = (__bf16*)wp; wp += (size_t)640 * 512 * 2;
  __bf16* pk_c2   = (__bf16*)wp; wp += (size_t)20  * 512 * 2;
  __bf16* pk_c3   = (__bf16*)wp; wp += (size_t)48  * 512 * 2;
  __bf16* pk_e1   = (__bf16*)wp; wp += (size_t)128 * 512 * 2;
  __bf16* pk_e2   = (__bf16*)wp; wp += (size_t)128 * 512 * 2;
  __bf16* pk_u1   = (__bf16*)wp; wp += (size_t)64  * 512 * 2;

  auto pack = [&](const float* src, __bf16* dst, int frags, int layer_stride,
                  int fpl, int ktiles, int ntiles, int ocs, int ks) {
    int threads = frags * 32;
    k_packB<<<(threads + 255) / 256, 256, 0, stream>>>(src, dst, frags, layer_stride,
                                                       fpl, ktiles, ntiles, ocs, ks);
  };
  // frag order per layer: ((tap*ktiles + kt)*ntiles + nt)
  pack(dil_w,   pk_dil,  160, 2048, 4,  1, 2,  64,  2);   // taps=2
  pack(condc_w, pk_cond, 320, 4096, 8,  4, 2,  128, 1);
  pack(filt_w,  pk_filt, 80,  1024, 2,  1, 2,  32,  1);
  pack(gate_w,  pk_gate, 80,  1024, 2,  1, 2,  32,  1);
  pack(res_w,   pk_res,  80,  1024, 2,  1, 2,  32,  1);
  pack(skip_w,  pk_skip, 640, 8192, 16, 1, 16, 32,  1);
  pack(cond_w2, pk_c2,   20,  0,    20, 1, 4,  160, 5);   // taps=5
  pack(cond_w3, pk_c3,   48,  0,    48, 2, 8,  192, 3);   // taps=3
  pack(end1_w,  pk_e1,   128, 0,    128,8, 16, 256, 1);
  pack(end2_w,  pk_e2,   128, 0,    128,8, 16, 256, 1);
  pack(unc_w1,  pk_u1,   64,  0,    64, 8, 8,  256, 1);

  const int ptBlocks   = (int)(bt / 256);   // one thread per (b,t)
  const int tileWaves  = (int)(bt / 16);    // one wave per 16-time tile
  const int tileBlocks = tileWaves * 32 / 256;

  k_cond1<<<ptBlocks, 256, 0, stream>>>(x, cond_w1, cond_b1, cond_bn1, c1);
  k_cond2<<<tileBlocks, 256, 0, stream>>>(c1, pk_c2, cond_b2, cond_bn2, c2);
  k_cond3<<<tileBlocks, 256, 0, stream>>>(c2, pk_c3, cond_b3, cond_bn3, c3);
  k_start<<<ptBlocks, 256, 0, stream>>>(x, start_w, start_b, h0);

  for (int i = 0; i < NLAYERS; ++i) {
    const float* hi_ = (i & 1) ? h1 : h0;
    float*       ho_ = (i & 1) ? h0 : h1;
    int d = 1 << (i % 10);
    k_layer<<<tileBlocks, 256, 0, stream>>>(hi_, ho_, c3, gall + (size_t)i * bt * 32,
                                            pk_dil, dil_b, pk_cond, condc_b,
                                            pk_filt, filt_b, pk_gate, gate_b,
                                            pk_res, res_b, blk_bn, i, d);
  }

  k_headA<<<(int)(bt / 32), 128, 0, stream>>>(gall, pk_skip, skip_b, pk_e1, end1_b, y1);
  k_headB<<<tileWaves, 128, 0, stream>>>(y1, pk_e2, end2_b, pred_w, pred_b,
                                         pk_u1, unc_b1, unc_w2, unc_b2,
                                         (float*)d_out);
}
